// xLSTM_60163901882429
// MI455X (gfx1250) — matrix operand order repaired and checked by reading, not compiled, not run on hardware
//
#include <hip/hip_runtime.h>
#include <hip/hip_bf16.h>
#include <stdint.h>

#define B_ 32
#define T_ 512
#define D_ 1024
#define H_ 1024
#define R_ 64
#define G_ 64           // workgroups per scan kernel (grid-wide barrier members)
#define HS_ (H_ / G_)   // 16 H-columns owned per workgroup

typedef __bf16 bf16_t;
typedef __attribute__((ext_vector_type(8)))  __bf16 bf16x8;
typedef __attribute__((ext_vector_type(16))) __bf16 bf16x16;
typedef __attribute__((ext_vector_type(8)))  float  f32x8;
typedef __attribute__((ext_vector_type(4)))  uint32_t u32x4;
typedef __attribute__((ext_vector_type(8)))  int      i32x8;
typedef __attribute__((ext_vector_type(4)))  int      i32x4;

// ---------- helpers ----------------------------------------------------------

__device__ __forceinline__ bf16_t f2bf(float x) {
  union { float f; uint32_t u; } v; v.f = x;
  uint32_t r = v.u + 0x7FFFu + ((v.u >> 16) & 1u);   // round-to-nearest-even
  uint16_t h = (uint16_t)(r >> 16);
  return __builtin_bit_cast(__bf16, h);
}

__device__ __forceinline__ float sigm(float x) { return 1.0f / (1.0f + __expf(-x)); }

// A fragment (16x32 bf16, row-major source, leading dim ld):
// lane L: row = row0 + L%16 ; elems 0..7 = K kk+(L/16)*8.. ; elems 8..15 = +16
__device__ __forceinline__ bf16x16 load_a_frag(const bf16_t* base, int ld, int row0,
                                               int kk, int lane) {
  const bf16_t* p = base + (size_t)(row0 + (lane & 15)) * ld + kk + ((lane >> 4) << 3);
  bf16x8 lo = *(const bf16x8*)p;
  bf16x8 hi = *(const bf16x8*)(p + 16);
  return __builtin_shufflevector(lo, hi, 0,1,2,3,4,5,6,7,8,9,10,11,12,13,14,15);
}

// B fragment (32x16 bf16). B[k][n] = W[col0+n][k] with W row-major (ld = K dim):
// lane L: col = col0 + L%16 ; elems 0..7 = K kk+(L/16)*8.. ; elems 8..15 = +16
__device__ __forceinline__ bf16x16 load_b_frag(const bf16_t* base, int ld, int col0,
                                               int kk, int lane) {
  const bf16_t* p = base + (size_t)(col0 + (lane & 15)) * ld + kk + ((lane >> 4) << 3);
  bf16x8 lo = *(const bf16x8*)p;
  bf16x8 hi = *(const bf16x8*)(p + 16);
  return __builtin_shufflevector(lo, hi, 0,1,2,3,4,5,6,7,8,9,10,11,12,13,14,15);
}

__device__ __forceinline__ f32x8 wmma_bf16(bf16x16 a, bf16x16 b, f32x8 c) {
  return __builtin_amdgcn_wmma_f32_16x16x32_bf16(false, a, false, b, (short)0, c,
                                                 false, false);
}

// Tensor Data Mover: DMA a 2-D bf16 tile [rows x width] (row stride = stride
// elems) from global into LDS at byte offset lds_addr. D# layout per
// CDNA5 ISA (group0: count/lds/global/type, group1:
// data_size=2B, tensor dims, tile dims, dim0 stride). Tracked by TENSORcnt.
// This toolchain exposes the 6-arg builtin (extra int32x8 group, zero-filled).
__device__ __forceinline__ void tdm_load_2d_bf16(uint32_t lds_addr,
                                                 const bf16_t* gsrc,
                                                 uint32_t width, uint32_t rows,
                                                 uint32_t stride) {
  uint64_t ga = (uint64_t)(uintptr_t)gsrc;
  u32x4 g0;
  g0.x = 1u;                                            // count=1, user mode
  g0.y = lds_addr;                                      // LDS byte address
  g0.z = (uint32_t)ga;                                  // global_addr[31:0]
  g0.w = (uint32_t)((ga >> 32) & 0x1FFFFFFu) | (2u << 30);  // addr[56:32]|type=2
  i32x8 g1;
  g1[0] = (int)(1u << 16);                              // data_size=1 -> 2 bytes
  g1[1] = (int)((width & 0xFFFFu) << 16);               // tensor_dim0[15:0]
  g1[2] = (int)((width >> 16) | ((rows & 0xFFFFu) << 16));   // dim0 hi | dim1 lo
  g1[3] = (int)((rows >> 16) | ((width & 0xFFFFu) << 16));   // dim1 hi | tile_dim0
  g1[4] = (int)(rows & 0xFFFFu);                        // tile_dim1 ; tile_dim2=0
  g1[5] = (int)stride;                                  // tensor_dim0_stride lo32
  g1[6] = 0;                                            // stride hi | dim1_stride lo
  g1[7] = 0;
  i32x4 gz4 = {0, 0, 0, 0};                             // 2-D: groups 2/3 unused
  i32x8 gz8 = {0, 0, 0, 0, 0, 0, 0, 0};
  __builtin_amdgcn_tensor_load_to_lds(g0, g1, gz4, gz4, gz8, 0);
}

// device-wide barrier across G_ resident workgroups (sense-reversing, agent scope)
__device__ __forceinline__ void grid_barrier(unsigned* cnt, unsigned* gen, unsigned nb) {
  __syncthreads();
  if (threadIdx.x == 0) {
    __threadfence();
    unsigned g = __hip_atomic_load(gen, __ATOMIC_RELAXED, __HIP_MEMORY_SCOPE_AGENT);
    unsigned a = __hip_atomic_fetch_add(cnt, 1u, __ATOMIC_ACQ_REL, __HIP_MEMORY_SCOPE_AGENT);
    if (a == nb - 1u) {
      __hip_atomic_store(cnt, 0u, __ATOMIC_RELAXED, __HIP_MEMORY_SCOPE_AGENT);
      __hip_atomic_fetch_add(gen, 1u, __ATOMIC_RELEASE, __HIP_MEMORY_SCOPE_AGENT);
    } else {
      while (__hip_atomic_load(gen, __ATOMIC_ACQUIRE, __HIP_MEMORY_SCOPE_AGENT) == g)
        __builtin_amdgcn_s_sleep(2);
    }
    __threadfence();
  }
  __syncthreads();
}

// ---------- utility kernels --------------------------------------------------

__global__ __launch_bounds__(256) void cvt_f32_bf16(const float* __restrict__ in,
                                                    bf16_t* __restrict__ out, size_t n) {
  size_t i = (size_t)blockIdx.x * blockDim.x + threadIdx.x;
  size_t s = (size_t)gridDim.x * blockDim.x;
  for (; i < n; i += s) out[i] = f2bf(in[i]);
}

__global__ __launch_bounds__(256) void zero_u32(unsigned* __restrict__ p, size_t n) {
  size_t i = (size_t)blockIdx.x * blockDim.x + threadIdx.x;
  size_t s = (size_t)gridDim.x * blockDim.x;
  for (; i < n; i += s) p[i] = 0u;
}

// ---------- batched GEMM: Y[M,N] = Xb[M,K] @ Wb[N,K]^T (+bias) ---------------
// block = 8 waves -> 64x128 tile ; each wave: 32x32 via 2x2 WMMA (frag reuse)

__global__ __launch_bounds__(256) void gemm_bf16(const bf16_t* __restrict__ Xb,
                                                 const bf16_t* __restrict__ Wb,
                                                 const float* __restrict__ bias,
                                                 float* __restrict__ Y,
                                                 int M, int N, int K) {
  int tid = threadIdx.x, lane = tid & 31, wave = tid >> 5;
  int m0 = blockIdx.y * 64 + (wave >> 2) * 32;
  int n0 = blockIdx.x * 128 + (wave & 3) * 32;
  if (n0 >= N || m0 >= M) return;            // wave-uniform
  f32x8 acc[2][2] = {};
  for (int kk = 0; kk < K; kk += 32) {
    if (kk + 128 < K) {                       // gfx1250 global_prefetch_b8
      __builtin_prefetch(Xb + (size_t)(m0 + (lane & 15)) * K + kk + 128, 0, 3);
      __builtin_prefetch(Wb + (size_t)(n0 + (lane & 15)) * K + kk + 128, 0, 3);
    }
    bf16x16 a0 = load_a_frag(Xb, K, m0, kk, lane);
    bf16x16 a1 = load_a_frag(Xb, K, m0 + 16, kk, lane);
    bf16x16 b0 = load_b_frag(Wb, K, n0, kk, lane);
    bf16x16 b1 = load_b_frag(Wb, K, n0 + 16, kk, lane);
    acc[0][0] = wmma_bf16(a0, b0, acc[0][0]);
    acc[0][1] = wmma_bf16(a0, b1, acc[0][1]);
    acc[1][0] = wmma_bf16(a1, b0, acc[1][0]);
    acc[1][1] = wmma_bf16(a1, b1, acc[1][1]);
  }
#pragma unroll
  for (int i = 0; i < 2; ++i)
#pragma unroll
    for (int j = 0; j < 2; ++j) {
      int col = n0 + j * 16 + (lane & 15);
      float bv = bias ? bias[col] : 0.0f;
      int mb = m0 + i * 16 + ((lane >> 4) << 3);
#pragma unroll
      for (int v = 0; v < 8; ++v)
        Y[(size_t)(mb + v) * N + col] = acc[i][j][v] + bv;
    }
}

// ---------- sLSTM persistent scan -------------------------------------------
// Each of 64 WGs owns HS_=16 columns of H; its 4*16 rows of Us (bf16) are
// DMA'd into LDS by the Tensor Data Mover (4 descriptors, one per gate block).
// Per step: Z[32,64] = h_prev[32,1024] @ Uslice^T (WMMA, B from LDS), then gates.

__global__ __launch_bounds__(256) void slstm_scan(const float* __restrict__ xw,
                                                  const bf16_t* __restrict__ Usb,
                                                  const float* __restrict__ bUs,
                                                  const float* __restrict__ alpha,
                                                  bf16_t* __restrict__ hA,
                                                  bf16_t* __restrict__ hB,
                                                  float* __restrict__ cs,
                                                  bf16_t* __restrict__ outs,
                                                  unsigned* __restrict__ sync) {
  __shared__ bf16_t Ulds[4 * HS_][H_];   // 128 KB: rows = [i|f|o|g] x 16 slice cols
  __shared__ float  Z[B_][4 * HS_];      // 8 KB
  int tid = threadIdx.x, lane = tid & 31, wave = tid >> 5;
  int h0 = blockIdx.x * HS_;

  // TDM: stage the 4 gate blocks of the U slice (16x1024 bf16 tiles) into LDS
  if (wave < 4) {
    int q = wave;
    tdm_load_2d_bf16((uint32_t)(uintptr_t)(void*)&Ulds[q * HS_][0],
                     Usb + (size_t)(q * H_ + h0) * H_, H_, HS_, H_);
    __builtin_amdgcn_s_wait_tensorcnt(0);
  }
  __syncthreads();

  for (int t = 0; t < T_; ++t) {
    const bf16_t* hp = (t & 1) ? hB : hA;
    bf16_t*       hn = (t & 1) ? hA : hB;
    {   // recurrent GEMM slice: 2 (M) x 4 (N) tiles over 8 waves, K=1024
      int mt = wave >> 2, nt = wave & 3;
      f32x8 acc = {};
      for (int kk = 0; kk < H_; kk += 32) {
        bf16x16 a = load_a_frag(hp, H_, mt * 16, kk, lane);
        bf16x16 b = load_b_frag(&Ulds[0][0], H_, nt * 16, kk, lane);  // ds_load_b128
        acc = wmma_bf16(a, b, acc);
      }
      int n = nt * 16 + (lane & 15);
      int mb = mt * 16 + ((lane >> 4) << 3);
#pragma unroll
      for (int v = 0; v < 8; ++v) Z[mb + v][n] = acc[v];
    }
    __syncthreads();
    // gates + state update for the owned 16 columns
    for (int e = tid; e < B_ * HS_; e += 256) {
      int b = e >> 4, j = e & 15;
      int col = h0 + j;
      size_t xr = (size_t)(b * T_ + t) * (4 * H_);
      float zi = Z[b][j]          + xw[xr + col]            + bUs[col];
      float zf = Z[b][HS_ + j]    + xw[xr + H_ + col]       + bUs[H_ + col];
      float zo = Z[b][2*HS_ + j]  + xw[xr + 2 * H_ + col]   + bUs[2 * H_ + col];
      float zg = Z[b][3*HS_ + j]  + xw[xr + 3 * H_ + col]   + bUs[3 * H_ + col];
      float ig = sigm(zi), fg = sigm(zf), og = sigm(zo), gg = tanhf(zg);
      float cc = alpha[col] * (fg * cs[b * H_ + col] + ig * gg);
      cs[b * H_ + col] = cc;
      float hh = og * tanhf(cc);
      hn[b * H_ + col] = f2bf(hh);
      outs[(size_t)(b * T_ + t) * H_ + col] = f2bf(hh);
    }
    grid_barrier(sync, sync + 1, G_);
  }
}

// ---------- mLSTM persistent scan -------------------------------------------
// Adds low-rank mix: q = xa_t * (h @ Bm^T) ; mix = q @ P_slice^T (all WMMA).
// Bm (64x1024 bf16) is TDM-DMA'd into LDS once; U slice streamed from the
// L2-resident bf16 copy (8 MB total across the device).

__global__ __launch_bounds__(256) void mlstm_scan(const float* __restrict__ xw,
                                                  const float* __restrict__ xa,
                                                  const bf16_t* __restrict__ Umb,
                                                  const bf16_t* __restrict__ Bmb,
                                                  const float* __restrict__ P,
                                                  const float* __restrict__ bUm,
                                                  bf16_t* __restrict__ hA,
                                                  bf16_t* __restrict__ hB,
                                                  float* __restrict__ cs,
                                                  float* __restrict__ out,
                                                  unsigned* __restrict__ sync) {
  __shared__ bf16_t Bmlds[R_][H_];       // 128 KB (TDM-loaded)
  __shared__ float  Z[B_][4 * HS_];      // 8 KB
  __shared__ float  qf[B_][R_];          // 8 KB
  __shared__ bf16_t qb[B_][R_];          // 4 KB
  __shared__ bf16_t Plds[HS_][R_];       // 2 KB
  __shared__ float  mixs[B_][HS_];       // 2 KB
  int tid = threadIdx.x, lane = tid & 31, wave = tid >> 5;
  int h0 = blockIdx.x * HS_;

  if (wave == 0) {    // TDM: whole Bm (64x1024 bf16) -> LDS
    tdm_load_2d_bf16((uint32_t)(uintptr_t)(void*)&Bmlds[0][0], Bmb, H_, R_, H_);
    __builtin_amdgcn_s_wait_tensorcnt(0);
  }
  for (int i = tid; i < HS_ * R_; i += 256) {
    int j = i >> 6, r = i & 63;
    Plds[j][r] = f2bf(P[(size_t)(h0 + j) * R_ + r]);
  }
  __syncthreads();

  for (int t = 0; t < T_; ++t) {
    const bf16_t* hp = (t & 1) ? hB : hA;
    bf16_t*       hn = (t & 1) ? hA : hB;
    int mt = wave >> 2, nt = wave & 3;
    {   // Z slice: h_prev @ Uslice^T   (B fragments from L2-resident Umb)
      const bf16_t* Ub = Umb + (size_t)(nt * H_ + h0) * H_;
      f32x8 acc = {};
      for (int kk = 0; kk < H_; kk += 32) {
        bf16x16 a = load_a_frag(hp, H_, mt * 16, kk, lane);
        bf16x16 b = load_b_frag(Ub, H_, 0, kk, lane);
        acc = wmma_bf16(a, b, acc);
      }
      int n = nt * 16 + (lane & 15);
      int mb = mt * 16 + ((lane >> 4) << 3);
#pragma unroll
      for (int v = 0; v < 8; ++v) Z[mb + v][n] = acc[v];
    }
    {   // qf = h_prev @ Bm^T -> [32, 64]   (B fragments from LDS)
      f32x8 acc = {};
      for (int kk = 0; kk < H_; kk += 32) {
        bf16x16 a = load_a_frag(hp, H_, mt * 16, kk, lane);
        bf16x16 b = load_b_frag(&Bmlds[0][0], H_, nt * 16, kk, lane);
        acc = wmma_bf16(a, b, acc);
      }
      int n = nt * 16 + (lane & 15);
      int mb = mt * 16 + ((lane >> 4) << 3);
#pragma unroll
      for (int v = 0; v < 8; ++v) qf[mb + v][n] = acc[v];
    }
    __syncthreads();
    for (int e = tid; e < B_ * R_; e += 256) {   // q = xa_t * qf, to bf16
      int b = e >> 6, r = e & 63;
      qb[b][r] = f2bf(qf[b][r] * xa[(size_t)(b * T_ + t) * R_ + r]);
    }
    __syncthreads();
    if (wave < 2) {   // mix = q @ Pslice^T : [32,64] x [64,16], K=64
      f32x8 acc = {};
      for (int kk = 0; kk < R_; kk += 32) {
        bf16x16 a = load_a_frag(&qb[0][0], R_, wave * 16, kk, lane);
        bf16x16 b = load_b_frag(&Plds[0][0], R_, 0, kk, lane);
        acc = wmma_bf16(a, b, acc);
      }
      int n = lane & 15;
      int mb = wave * 16 + ((lane >> 4) << 3);
#pragma unroll
      for (int v = 0; v < 8; ++v) mixs[mb + v][n] = acc[v];
    }
    __syncthreads();
    for (int e = tid; e < B_ * HS_; e += 256) {
      int b = e >> 4, j = e & 15;
      int col = h0 + j;
      size_t xr = (size_t)(b * T_ + t) * (4 * H_);
      float zi = Z[b][j]          + xw[xr + col]            + bUm[col];
      float zf = Z[b][HS_ + j]    + xw[xr + H_ + col]       + bUm[H_ + col];
      float zo = Z[b][2*HS_ + j]  + xw[xr + 2 * H_ + col]   + bUm[2 * H_ + col];
      float zg = Z[b][3*HS_ + j]  + xw[xr + 3 * H_ + col]   + bUm[3 * H_ + col];
      float ig = sigm(zi), fg = sigm(zf), og = sigm(zo), gg = tanhf(zg);
      float cc = fg * cs[b * H_ + col] + ig * gg + 0.1f * mixs[b][j];
      cs[b * H_ + col] = cc;
      float hh = og * tanhf(cc);
      hn[b * H_ + col] = f2bf(hh);
      out[(size_t)(b * T_ + t) * H_ + col] = hh;
    }
    grid_barrier(sync, sync + 1, G_);
  }
}

// ---------- host launcher ----------------------------------------------------

static constexpr size_t aln(size_t x) { return (x + 255) & ~(size_t)255; }

extern "C" void kernel_launch(void* const* d_in, const int* in_sizes, int n_in,
                              void* d_out, int out_size, void* d_ws, size_t ws_size,
                              hipStream_t stream) {
  (void)in_sizes; (void)n_in; (void)out_size; (void)ws_size;
  const float* x    = (const float*)d_in[0];
  const float* Ws   = (const float*)d_in[1];
  const float* bWs  = (const float*)d_in[2];
  const float* Us   = (const float*)d_in[3];
  const float* bUs  = (const float*)d_in[4];
  const float* alph = (const float*)d_in[5];
  const float* Wm   = (const float*)d_in[6];
  const float* bWm  = (const float*)d_in[7];
  const float* Um   = (const float*)d_in[8];
  const float* bUm  = (const float*)d_in[9];
  const float* A    = (const float*)d_in[10];
  const float* Bm   = (const float*)d_in[11];
  const float* P    = (const float*)d_in[12];
  float* out = (float*)d_out;
  char*  w   = (char*)d_ws;

  // workspace layout
  constexpr size_t SZ_H = (size_t)B_ * H_ * sizeof(bf16_t);   // 64 KB
  constexpr size_t SZ_C = (size_t)B_ * H_ * sizeof(float);    // 128 KB
  constexpr size_t OFF_HA   = 256;                            // sync at 0
  constexpr size_t OFF_HB   = OFF_HA + SZ_H;
  constexpr size_t OFF_CS   = OFF_HB + SZ_H;
  constexpr size_t STATE_BYTES = OFF_CS + SZ_C;
  constexpr size_t OFF_XBF  = aln(STATE_BYTES);
  constexpr size_t OFF_WSBF = aln(OFF_XBF  + (size_t)B_ * T_ * D_ * 2);
  constexpr size_t OFF_USBF = aln(OFF_WSBF + (size_t)4 * H_ * D_ * 2);
  constexpr size_t OFF_WMBF = aln(OFF_USBF + (size_t)4 * H_ * H_ * 2);
  constexpr size_t OFF_UMBF = aln(OFF_WMBF + (size_t)4 * H_ * H_ * 2);
  constexpr size_t OFF_ABF  = aln(OFF_UMBF + (size_t)4 * H_ * H_ * 2);
  constexpr size_t OFF_BMBF = aln(OFF_ABF  + (size_t)R_ * H_ * 2);
  constexpr size_t OFF_OUTS = aln(OFF_BMBF + (size_t)R_ * H_ * 2);
  constexpr size_t OFF_XW   = aln(OFF_OUTS + (size_t)B_ * T_ * H_ * 2);
  constexpr size_t OFF_XA   = aln(OFF_XW   + (size_t)B_ * T_ * 4 * H_ * 4);

  unsigned* syncp = (unsigned*)w;
  bf16_t* hA    = (bf16_t*)(w + OFF_HA);
  bf16_t* hB    = (bf16_t*)(w + OFF_HB);
  float*  cst   = (float*) (w + OFF_CS);
  bf16_t* xbf   = (bf16_t*)(w + OFF_XBF);
  bf16_t* wsbf  = (bf16_t*)(w + OFF_WSBF);
  bf16_t* usbf  = (bf16_t*)(w + OFF_USBF);
  bf16_t* wmbf  = (bf16_t*)(w + OFF_WMBF);
  bf16_t* umbf  = (bf16_t*)(w + OFF_UMBF);
  bf16_t* abf   = (bf16_t*)(w + OFF_ABF);
  bf16_t* bmbf  = (bf16_t*)(w + OFF_BMBF);
  bf16_t* outsb = (bf16_t*)(w + OFF_OUTS);
  float*  xw    = (float*) (w + OFF_XW);
  float*  xa    = (float*) (w + OFF_XA);

  // 1) fp32 -> bf16 staging
  cvt_f32_bf16<<<4096, 256, 0, stream>>>(x,  xbf,  (size_t)B_ * T_ * D_);
  cvt_f32_bf16<<<2048, 256, 0, stream>>>(Ws, wsbf, (size_t)4 * H_ * D_);
  cvt_f32_bf16<<<2048, 256, 0, stream>>>(Us, usbf, (size_t)4 * H_ * H_);
  cvt_f32_bf16<<<2048, 256, 0, stream>>>(Wm, wmbf, (size_t)4 * H_ * H_);
  cvt_f32_bf16<<<2048, 256, 0, stream>>>(Um, umbf, (size_t)4 * H_ * H_);
  cvt_f32_bf16<<<64,   256, 0, stream>>>(A,  abf,  (size_t)R_ * H_);
  cvt_f32_bf16<<<64,   256, 0, stream>>>(Bm, bmbf, (size_t)R_ * H_);

  const int M = B_ * T_;   // 16384 rows, row (b*T + t)

  // 2) sLSTM input projection: xw = x @ Ws^T + bWs  -> [M, 4H] fp32
  gemm_bf16<<<dim3((4 * H_) / 128, M / 64), 256, 0, stream>>>(
      xbf, wsbf, bWs, xw, M, 4 * H_, D_);

  // 3) sLSTM scan (persistent, 64 WGs, TDM-staged LDS weights, WMMA recurrence)
  zero_u32<<<64, 256, 0, stream>>>((unsigned*)w, STATE_BYTES / 4);
  slstm_scan<<<G_, 256, 0, stream>>>(xw, usbf, bUs, alph, hA, hB, cst, outsb, syncp);

  // 4) mLSTM input projections: xw = out_s @ Wm^T + bWm ; xa = out_s @ A^T
  gemm_bf16<<<dim3((4 * H_) / 128, M / 64), 256, 0, stream>>>(
      outsb, wmbf, bWm, xw, M, 4 * H_, H_);
  gemm_bf16<<<dim3(1, M / 64), 256, 0, stream>>>(
      outsb, abf, (const float*)nullptr, xa, M, R_, H_);

  // 5) mLSTM scan -> final output [B, T, H] fp32
  zero_u32<<<64, 256, 0, stream>>>((unsigned*)w, STATE_BYTES / 4);
  mlstm_scan<<<G_, 256, 0, stream>>>(xw, xa, umbf, bmbf, P, bUm, hA, hB, cst,
                                     out, syncp);
}
